// GPT_61409442398424
// MI455X (gfx1250) — matrix operand order repaired and checked by reading, not compiled, not run on hardware
//
#include <hip/hip_runtime.h>
#include <cstdint>

// ---- model constants (match reference) ----
#define B_     2
#define S_     1024
#define DM_    1024
#define H_     16
#define HKV_   4
#define L_     4
#define DH_    64
#define HASH_  16384
#define EPS_   1.1920929e-07f
#define THRESH_ 0.8f
#define DECAY_  0.9f

typedef __attribute__((ext_vector_type(16))) __bf16 v16bf;
typedef __attribute__((ext_vector_type(8)))  __bf16 v8bf;
typedef __attribute__((ext_vector_type(8)))  float  v8f;

static __device__ __forceinline__ __bf16 f2bf(float f) {
  union { float f; unsigned u; } a; a.f = f;
  unsigned r = (a.u + 0x7FFFu + ((a.u >> 16) & 1u)) >> 16;   // RNE
  union { unsigned short s; __bf16 b; } o; o.s = (unsigned short)r;
  return o.b;
}
static __device__ __forceinline__ unsigned pack2(float a, float b) {
  union { __bf16 h[2]; unsigned u; } x; x.h[0] = f2bf(a); x.h[1] = f2bf(b); return x.u;
}
static __device__ __forceinline__ v16bf cat8(v8bf lo, v8bf hi) {
  v16bf r;
#pragma unroll
  for (int i = 0; i < 8; ++i) { r[i] = lo[i]; r[i + 8] = hi[i]; }
  return r;
}
static __device__ __forceinline__ v8f v8f_zero() {
  v8f r;
#pragma unroll
  for (int i = 0; i < 8; ++i) r[i] = 0.f;
  return r;
}

// =====================================================================
// Tiled GEMM: C[M,N] = A[M,K] * W[N,K]^T  (both K-contiguous, fp32 in,
// bf16 WMMA compute, fp32 accumulate).  MODE 0: plain, 1: leaky_relu2,
// 2: C = resid + scale[col]*acc (residual add, scale broadcast over N).
// Requires M%64==0, N%128==0, K%32==0 (true for all shapes here).
// =====================================================================
#define BM 64
#define BN 128
#define BK 32
#define LDT 40   // padded LDS row stride (bf16 elems)

template<int MODE>
__global__ __launch_bounds__(256) void gemm_bf16_wmma(
    const float* __restrict__ A, const float* __restrict__ W,
    float* __restrict__ C, int M, int N, int K,
    const float* __restrict__ resid, const float* __restrict__ scale) {
  __shared__ __bf16 As[BM * LDT];
  __shared__ __bf16 Ws[BN * LDT];

  const int tid  = threadIdx.x;
  const int lane = tid & 31;
  const int wv   = tid >> 5;
  const int hl   = lane >> 4;   // which 16-lane half
  const int lm   = lane & 15;
  const int wm   = wv & 1;      // 2 row groups of 32
  const int wn   = wv >> 1;     // 4 col groups of 32
  const int bm   = blockIdx.y * BM;
  const int bn   = blockIdx.x * BN;

  v8f acc[2][2];
#pragma unroll
  for (int i = 0; i < 2; ++i)
#pragma unroll
    for (int j = 0; j < 2; ++j) acc[i][j] = v8f_zero();

  for (int k0 = 0; k0 < K; k0 += BK) {
    // cooperative load + f32->bf16 convert: A tile 64x32, W tile 128x32
#pragma unroll
    for (int it = 0; it < 2; ++it) {
      int idx = tid + it * 256;
      int r = idx >> 3, c4 = idx & 7;
      const float4 f = *(const float4*)(A + (size_t)(bm + r) * K + k0 + c4 * 4);
      *(unsigned*)&As[r * LDT + c4 * 4]     = pack2(f.x, f.y);
      *(unsigned*)&As[r * LDT + c4 * 4 + 2] = pack2(f.z, f.w);
    }
#pragma unroll
    for (int it = 0; it < 4; ++it) {
      int idx = tid + it * 256;
      int r = idx >> 3, c4 = idx & 7;
      const float4 f = *(const float4*)(W + (size_t)(bn + r) * K + k0 + c4 * 4);
      *(unsigned*)&Ws[r * LDT + c4 * 4]     = pack2(f.x, f.y);
      *(unsigned*)&Ws[r * LDT + c4 * 4 + 2] = pack2(f.z, f.w);
    }
    __syncthreads();

    if (k0 + BK < K) {  // prefetch next K-slab (global_prefetch_b8)
      __builtin_prefetch(A + (size_t)(bm + (tid >> 2)) * K + k0 + BK, 0, 1);
      __builtin_prefetch(W + (size_t)(bn + (tid >> 1)) * K + k0 + BK, 0, 1);
    }

    v16bf af[2], bfr[2];
#pragma unroll
    for (int tm = 0; tm < 2; ++tm) {
      int ar = wm * 32 + tm * 16 + lm;
      v8bf lo = *(const v8bf*)&As[ar * LDT + hl * 8];          // K 0..7 / 8..15
      v8bf hi = *(const v8bf*)&As[ar * LDT + 16 + hl * 8];     // K 16..23 / 24..31
      af[tm] = cat8(lo, hi);
    }
#pragma unroll
    for (int tn = 0; tn < 2; ++tn) {
      int wr = wn * 32 + tn * 16 + lm;
      v8bf lo = *(const v8bf*)&Ws[wr * LDT + hl * 8];          // K 0..7 / 8..15
      v8bf hi = *(const v8bf*)&Ws[wr * LDT + 16 + hl * 8];     // K 16..23 / 24..31
      bfr[tn] = cat8(lo, hi);
    }
#pragma unroll
    for (int tm = 0; tm < 2; ++tm)
#pragma unroll
      for (int tn = 0; tn < 2; ++tn)
        acc[tm][tn] = __builtin_amdgcn_wmma_f32_16x16x32_bf16(
            false, af[tm], false, bfr[tn], (short)0, acc[tm][tn], false, false);
    __syncthreads();
  }

  // epilogue (C/D layout: lane<16 -> rows 0..7, lane>=16 -> rows 8..15)
#pragma unroll
  for (int tm = 0; tm < 2; ++tm)
#pragma unroll
    for (int tn = 0; tn < 2; ++tn) {
      int row0 = bm + wm * 32 + tm * 16 + hl * 8;
      int col  = bn + wn * 32 + tn * 16 + lm;
#pragma unroll
      for (int r = 0; r < 8; ++r) {
        size_t idx = (size_t)(row0 + r) * N + col;
        float vl = acc[tm][tn][r];
        if (MODE == 1) vl = vl > 0.f ? vl * vl : 0.01f * vl;
        if (MODE == 2) vl = resid[idx] + scale[col] * vl;
        C[idx] = vl;
      }
    }
}

// =====================================================================
// Embedding: x = uni[id % HASH] + bi[(prev*31+id) % HASH];  x0 = x
// =====================================================================
__global__ __launch_bounds__(256) void embed_kernel(
    const int* __restrict__ ids, const float* __restrict__ uni,
    const float* __restrict__ bi, float* __restrict__ x, float* __restrict__ x0) {
  int row = blockIdx.x;               // b*S + s
  int s   = row & (S_ - 1);
  int id  = ids[row];
  int pv  = (s == 0) ? 0 : ids[row - 1];
  int h1  = id & (HASH_ - 1);
  int h2  = (pv * 31 + id) & (HASH_ - 1);
  const float* u = uni + (size_t)h1 * DM_;
  const float* b = bi  + (size_t)h2 * DM_;
#pragma unroll
  for (int i = 0; i < 4; ++i) {
    int c = threadIdx.x + i * 256;
    float vv = u[c] + b[c];
    x [(size_t)row * DM_ + c] = vv;
    x0[(size_t)row * DM_ + c] = vv;
  }
}

// =====================================================================
// Residual mix (optional) + rmsnorm over DM.  MIX: x = rm0*x + rm1*x0
// (x updated in place), xn = rmsnorm(x).
// =====================================================================
template<bool MIX>
__global__ __launch_bounds__(256) void residnorm_kernel(
    float* __restrict__ x, const float* __restrict__ x0,
    float* __restrict__ xn, const float* __restrict__ rmx) {
  __shared__ float red[256];
  int row = blockIdx.x;
  float vals[4]; float ss = 0.f;
#pragma unroll
  for (int i = 0; i < 4; ++i) {
    int c = threadIdx.x + i * 256;
    float vv = x[(size_t)row * DM_ + c];
    if (MIX) {
      vv = rmx[c] * vv + rmx[DM_ + c] * x0[(size_t)row * DM_ + c];
      x[(size_t)row * DM_ + c] = vv;
    }
    vals[i] = vv; ss += vv * vv;
  }
  red[threadIdx.x] = ss;
  __syncthreads();
  for (int off = 128; off > 0; off >>= 1) {
    if (threadIdx.x < off) red[threadIdx.x] += red[threadIdx.x + off];
    __syncthreads();
  }
  float r = rsqrtf(red[0] / (float)DM_ + EPS_);
#pragma unroll
  for (int i = 0; i < 4; ++i) {
    int c = threadIdx.x + i * 256;
    xn[(size_t)row * DM_ + c] = vals[i] * r;
  }
}

// =====================================================================
// Per-(b,s,head) rmsnorm over DH then RoPE, in place, for q and k heads.
// Block = 64 threads (one head vector). blockIdx: (b*S+s)*(H+HKV)+hh.
// =====================================================================
__global__ __launch_bounds__(64) void qkprep_kernel(
    float* __restrict__ q, float* __restrict__ k) {
  __shared__ float red[64];
  __shared__ float nv[64];
  int bs = blockIdx.x / (H_ + HKV_);
  int hh = blockIdx.x % (H_ + HKV_);
  int s  = bs & (S_ - 1);
  float* p;
  if (hh < H_) p = q + ((size_t)bs * H_  + hh)        * DH_;
  else         p = k + ((size_t)bs * HKV_ + (hh - H_)) * DH_;
  int d = threadIdx.x;
  float vv = p[d];
  red[d] = vv * vv;
  __syncthreads();
  for (int off = 32; off > 0; off >>= 1) {
    if (d < off) red[d] += red[d + off];
    __syncthreads();
  }
  float r = rsqrtf(red[0] / (float)DH_ + EPS_);
  nv[d] = vv * r;
  __syncthreads();
  if (d < 32) {
    float fr = (float)s * __powf(10000.f, -(float)d / 32.f);
    float sn, cs;
    __sincosf(fr, &sn, &cs);
    float x1 = nv[d], x2 = nv[d + 32];
    p[d]      =  x1 * cs + x2 * sn;
    p[d + 32] = -x1 * sn + x2 * cs;
  }
}

// =====================================================================
// LIF spiking gate (sequential over S) + q_gain: q = q * spike * gain
// One thread per (b,h,d) chain: 2048 threads total.
// =====================================================================
__global__ __launch_bounds__(256) void lif_kernel(
    float* __restrict__ q, const float* __restrict__ qg) {
  int g = blockIdx.x * 256 + threadIdx.x;
  int b   = g / (H_ * DH_);
  int rem = g % (H_ * DH_);
  int h   = rem / DH_;
  int d   = rem % DH_;
  float gain = qg[h];
  float mem = 0.f;
  for (int s = 0; s < S_; ++s) {
    size_t idx = (((size_t)b * S_ + s) * H_ + h) * DH_ + d;
    float vv = q[idx];
    mem = DECAY_ * mem + vv;
    float sp = (mem >= THRESH_) ? 1.f : 0.f;
    q[idx] = vv * sp * gain;
    mem *= (1.f - sp);
  }
}

// =====================================================================
// Flash attention (causal, GQA).  One block = (b,h) x 128 queries,
// 8 waves x 16 queries each.  QK^T and P*V both on bf16 WMMA; online
// softmax in registers via wave32 shfl_xor butterflies.
// Scale 1/sqrt(DH) folded into Q at bf16 conversion.
// =====================================================================
#define QB 128
#define TBK 64
#define LDK 72

__global__ __launch_bounds__(256) void flash_kernel(
    const float* __restrict__ q, const float* __restrict__ k,
    const float* __restrict__ v, float* __restrict__ y) {
  __shared__ __bf16 Ks[TBK * LDK];        // K tile, row = key t, d contiguous
  __shared__ __bf16 Vt[DH_ * LDK];        // V tile transposed: row = d, t contiguous
  __shared__ __bf16 Pb[8 * 16 * LDK];     // per-wave P tiles (16 x 64)

  const int nqb  = S_ / QB;
  const int bh   = blockIdx.x / nqb;
  const int qblk = blockIdx.x % nqb;
  const int b = bh / H_, h = bh % H_;
  const int kvh = h / (H_ / HKV_);
  const int q0 = qblk * QB;

  const int tid = threadIdx.x, lane = tid & 31, wv = tid >> 5;
  const int hl = lane >> 4, lm = lane & 15;
  const int qrow = q0 + wv * 16;

  // Q fragments (A-operand, per-lane row = qrow+lm), scaled by 0.125
  v16bf qa[2];
  {
    const float* qp = q + (((size_t)b * S_ + (qrow + lm)) * H_ + h) * DH_;
#pragma unroll
    for (int kk = 0; kk < 2; ++kk) {
      int g0 = kk * 32 + hl * 8;
      int g1 = kk * 32 + 16 + hl * 8;
      float4 f0 = *(const float4*)(qp + g0);
      float4 f1 = *(const float4*)(qp + g0 + 4);
      float4 f2 = *(const float4*)(qp + g1);
      float4 f3 = *(const float4*)(qp + g1 + 4);
      v16bf a;
      a[0]  = f2bf(f0.x * 0.125f); a[1]  = f2bf(f0.y * 0.125f);
      a[2]  = f2bf(f0.z * 0.125f); a[3]  = f2bf(f0.w * 0.125f);
      a[4]  = f2bf(f1.x * 0.125f); a[5]  = f2bf(f1.y * 0.125f);
      a[6]  = f2bf(f1.z * 0.125f); a[7]  = f2bf(f1.w * 0.125f);
      a[8]  = f2bf(f2.x * 0.125f); a[9]  = f2bf(f2.y * 0.125f);
      a[10] = f2bf(f2.z * 0.125f); a[11] = f2bf(f2.w * 0.125f);
      a[12] = f2bf(f3.x * 0.125f); a[13] = f2bf(f3.y * 0.125f);
      a[14] = f2bf(f3.z * 0.125f); a[15] = f2bf(f3.w * 0.125f);
      qa[kk] = a;
    }
  }

  v8f acc[4];
#pragma unroll
  for (int dt = 0; dt < 4; ++dt) acc[dt] = v8f_zero();
  float rm[8], rl[8];
#pragma unroll
  for (int r = 0; r < 8; ++r) { rm[r] = -1e30f; rl[r] = 0.f; }

  for (int t0 = 0; t0 < q0 + QB; t0 += TBK) {
    // cooperative K/V tile load (64 keys x 32 d), V stored transposed
#pragma unroll
    for (int it = 0; it < 2; ++it) {
      int idx = tid + it * 256;
      int t = idx >> 3, d4 = idx & 7;
      size_t base = (((size_t)b * S_ + t0 + t) * HKV_ + kvh) * DH_ + d4 * 4;
      const float4 f = *(const float4*)(k + base);
      *(unsigned*)&Ks[t * LDK + d4 * 4]     = pack2(f.x, f.y);
      *(unsigned*)&Ks[t * LDK + d4 * 4 + 2] = pack2(f.z, f.w);
      const float4 g = *(const float4*)(v + base);
      Vt[(d4 * 4 + 0) * LDK + t] = f2bf(g.x);
      Vt[(d4 * 4 + 1) * LDK + t] = f2bf(g.y);
      Vt[(d4 * 4 + 2) * LDK + t] = f2bf(g.z);
      Vt[(d4 * 4 + 3) * LDK + t] = f2bf(g.w);
    }
    __syncthreads();

    // S tile: 16 queries x 64 keys (4 WMMA col tiles x 2 K chunks)
    v8f st[4];
#pragma unroll
    for (int tn = 0; tn < 4; ++tn) st[tn] = v8f_zero();
#pragma unroll
    for (int tn = 0; tn < 4; ++tn) {
      int kr = tn * 16 + lm;
#pragma unroll
      for (int kk = 0; kk < 2; ++kk) {
        v8bf lo = *(const v8bf*)&Ks[kr * LDK + kk * 32 + hl * 8];
        v8bf hi = *(const v8bf*)&Ks[kr * LDK + kk * 32 + 16 + hl * 8];
        st[tn] = __builtin_amdgcn_wmma_f32_16x16x32_bf16(
            false, qa[kk], false, cat8(lo, hi), (short)0, st[tn], false, false);
      }
    }

    // online softmax (rows half*8+r live in this lane's half)
    float sc[8];
#pragma unroll
    for (int r = 0; r < 8; ++r) {
      int qg_ = qrow + hl * 8 + r;
      float mx = -1e30f;
      float pvv[4];
#pragma unroll
      for (int tn = 0; tn < 4; ++tn) {
        int t = t0 + tn * 16 + lm;
        float sv_ = (t <= qg_) ? st[tn][r] : -1e30f;
        pvv[tn] = sv_;
        mx = fmaxf(mx, sv_);
      }
#pragma unroll
      for (int m = 1; m < 16; m <<= 1) mx = fmaxf(mx, __shfl_xor(mx, m, 32));
      float mn  = fmaxf(rm[r], mx);
      float scl = __expf(rm[r] - mn);
      float ls  = 0.f;
#pragma unroll
      for (int tn = 0; tn < 4; ++tn) {
        int t = t0 + tn * 16 + lm;
        float p = (t <= qg_) ? __expf(pvv[tn] - mn) : 0.f;
        ls += p;
        Pb[(wv * 16 + hl * 8 + r) * LDK + tn * 16 + lm] = f2bf(p);
      }
#pragma unroll
      for (int m = 1; m < 16; m <<= 1) ls += __shfl_xor(ls, m, 32);
      rl[r] = rl[r] * scl + ls;
      rm[r] = mn;
      sc[r] = scl;
    }
#pragma unroll
    for (int dt = 0; dt < 4; ++dt)
#pragma unroll
      for (int r = 0; r < 8; ++r) acc[dt][r] *= sc[r];

    // P x V (contraction over t; Pb is wave-private -> in-wave LDS order)
#pragma unroll
    for (int kk = 0; kk < 2; ++kk) {
      v8bf plo = *(const v8bf*)&Pb[(wv * 16 + lm) * LDK + kk * 32 + hl * 8];
      v8bf phi = *(const v8bf*)&Pb[(wv * 16 + lm) * LDK + kk * 32 + 16 + hl * 8];
      v16bf pa = cat8(plo, phi);
#pragma unroll
      for (int dt = 0; dt < 4; ++dt) {
        int vr = dt * 16 + lm;
        v8bf blo = *(const v8bf*)&Vt[vr * LDK + kk * 32 + hl * 8];
        v8bf bhi = *(const v8bf*)&Vt[vr * LDK + kk * 32 + 16 + hl * 8];
        acc[dt] = __builtin_amdgcn_wmma_f32_16x16x32_bf16(
            false, pa, false, cat8(blo, bhi), (short)0, acc[dt], false, false);
      }
    }
    __syncthreads();
  }

  // finalize: y = acc / l, layout (b,s,h,d)
#pragma unroll
  for (int dt = 0; dt < 4; ++dt)
#pragma unroll
    for (int r = 0; r < 8; ++r) {
      int row = qrow + hl * 8 + r;
      int d   = dt * 16 + lm;
      y[(((size_t)b * S_ + row) * H_ + h) * DH_ + d] = acc[dt][r] / rl[r];
    }
}

// =====================================================================
// Host-side pipeline
// =====================================================================
extern "C" void kernel_launch(void* const* d_in, const int* in_sizes, int n_in,
                              void* d_out, int out_size, void* d_ws, size_t ws_size,
                              hipStream_t stream) {
  (void)in_sizes; (void)n_in; (void)out_size; (void)ws_size;
  const int*   ids = (const int*)  d_in[0];
  const float* uni = (const float*)d_in[1];
  const float* bi  = (const float*)d_in[2];
  const float* Wq  = (const float*)d_in[3];
  const float* Wk  = (const float*)d_in[4];
  const float* Wv  = (const float*)d_in[5];
  const float* Wo  = (const float*)d_in[6];
  const float* Wfc = (const float*)d_in[7];
  const float* Wp  = (const float*)d_in[8];
  const float* qg  = (const float*)d_in[9];
  const float* asc = (const float*)d_in[10];
  const float* msc = (const float*)d_in[11];
  const float* rmx = (const float*)d_in[12];
  float* out = (float*)d_out;

  float* ws = (float*)d_ws;
  const size_t MR = (size_t)B_ * S_;       // 2048 rows
  float* x  = ws;
  float* x0 = x  + MR * DM_;
  float* xn = x0 + MR * DM_;
  float* qb = xn + MR * DM_;
  float* yb = qb + MR * DM_;
  float* kb = yb + MR * DM_;
  float* vb = kb + MR * (HKV_ * DH_);
  float* hb = vb + MR * (HKV_ * DH_);

  const int M = (int)MR;
  dim3 blk(256);

  embed_kernel<<<M, blk, 0, stream>>>(ids, uni, bi, x, x0);

  for (int l = 0; l < L_; ++l) {
    residnorm_kernel<true><<<M, blk, 0, stream>>>(x, x0, xn,
                                                  rmx + (size_t)l * 2 * DM_);
    gemm_bf16_wmma<0><<<dim3(DM_ / BN, M / BM), blk, 0, stream>>>(
        xn, Wq + (size_t)l * DM_ * DM_, qb, M, DM_, DM_, nullptr, nullptr);
    gemm_bf16_wmma<0><<<dim3((HKV_ * DH_) / BN, M / BM), blk, 0, stream>>>(
        xn, Wk + (size_t)l * HKV_ * DH_ * DM_, kb, M, HKV_ * DH_, DM_, nullptr, nullptr);
    gemm_bf16_wmma<0><<<dim3((HKV_ * DH_) / BN, M / BM), blk, 0, stream>>>(
        xn, Wv + (size_t)l * HKV_ * DH_ * DM_, vb, M, HKV_ * DH_, DM_, nullptr, nullptr);
    qkprep_kernel<<<M * (H_ + HKV_), dim3(64), 0, stream>>>(qb, kb);
    lif_kernel<<<(B_ * H_ * DH_) / 256, blk, 0, stream>>>(qb, qg + (size_t)l * H_);
    flash_kernel<<<B_ * H_ * (S_ / QB), blk, 0, stream>>>(qb, kb, vb, yb);
    gemm_bf16_wmma<2><<<dim3(DM_ / BN, M / BM), blk, 0, stream>>>(
        yb, Wo + (size_t)l * DM_ * DM_, x, M, DM_, DM_, x, asc + (size_t)l * DM_);
    residnorm_kernel<false><<<M, blk, 0, stream>>>(x, nullptr, xn, nullptr);
    gemm_bf16_wmma<1><<<dim3((4 * DM_) / BN, M / BM), blk, 0, stream>>>(
        xn, Wfc + (size_t)l * 4 * DM_ * DM_, hb, M, 4 * DM_, DM_, nullptr, nullptr);
    gemm_bf16_wmma<2><<<dim3(DM_ / BN, M / BM), blk, 0, stream>>>(
        hb, Wp + (size_t)l * 4 * DM_ * DM_, x, M, DM_, 4 * DM_, x, msc + (size_t)l * DM_);
  }
  residnorm_kernel<false><<<M, blk, 0, stream>>>(x, nullptr, xn, nullptr);
  gemm_bf16_wmma<0><<<dim3(HASH_ / BN, M / BM), blk, 0, stream>>>(
      xn, uni, out, M, HASH_, DM_, nullptr, nullptr);
}
